// LSTMModel_40166534152868
// MI455X (gfx1250) — compile-verified, matrix operand order checked by reading, not run on hardware
//
#include <hip/hip_runtime.h>

// LSTM (H=16, IN=1) scan on gfx1250 using V_WMMA_F32_16X16X32_F16.
//
// Per 16-batch tile (one wave32 per tile), transposed formulation:
//   gates.T[g] = Whh_block[g] x h.T   via WMMA, K padded 16 -> 32.
// Pad K-slots carry input & bias so WMMA computes the FULL preactivation (C=0):
//   A[g][m][16] = W_ih[g*16+m],  A[g][m][17] = b_ih+b_hh   (half-0 lanes)
//   B[16][n]    = x_t[n],        B[17][n]    = 1.0
// Sigmoid gates (i,f,o) have their A block PRE-SCALED by 0.5 so that
//   sigmoid(z) = 0.5*tanh(z/2)+0.5 = fma(0.5, v_tanh(acc), 0.5)  -- no mul.
// D layout (f32 16x16): lane(n,half), vgpr r -> [M = r+8*half][N = n]
// B layout (f16 32x16): lane(n,half), elem e<8  -> [K = e+8*half][N = n]
//                                     elem e>=8 -> [K = 16+(e-8)+8*half][N = n]
// => h(new) (D layout) -> next-step B operand is LANE-LOCAL:
//    hB[r] = v_fma_mixlo_f16(og, tanh(c), 0); x_t enters via one fma_mixlo.

typedef __attribute__((ext_vector_type(16))) _Float16 v16h;
typedef __attribute__((ext_vector_type(8)))  float    v8f;

#define LSTM_H 16

#if __has_builtin(__builtin_amdgcn_tanhf)
__device__ __forceinline__ float tanh_fast(float x) {
  return __builtin_amdgcn_tanhf(x);                 // v_tanh_f32 (TRANS)
}
#else
__device__ __forceinline__ float tanh_fast(float x) {
  float e = __builtin_amdgcn_exp2f(-2.88539008177792681472f * x);
  return fmaf(2.0f, __builtin_amdgcn_rcpf(1.0f + e), -1.0f);
}
#endif

__global__ __launch_bounds__(256, 1)
void lstm16_wmma_kernel(const float* __restrict__ x,
                        const float* __restrict__ W_ih,
                        const float* __restrict__ W_hh,
                        const float* __restrict__ b_ih,
                        const float* __restrict__ b_hh,
                        const float* __restrict__ W_fc,
                        const float* __restrict__ b_fc,
                        float* __restrict__ out,
                        int B, int T)
{
  const int lane      = threadIdx.x & 31;
  const int waveInBlk = threadIdx.x >> 5;
  const int nTiles    = (B + 15) >> 4;

  int tile = blockIdx.x * (blockDim.x >> 5) + waveInBlk;   // wave-uniform
  if (tile >= nTiles) tile = nTiles - 1;                   // clamp, keep EXEC full
  int bbase = tile * 16;
  if (bbase + 16 > B) bbase = (B >= 16) ? (B - 16) : 0;    // tail dup, same result

  const int n    = lane & 15;   // N column = batch-in-tile; also A row m
  const int half = lane >> 4;   // +8 selector in M/K striping
  const float xsel = (half == 0) ? 1.0f : 0.0f;            // K=16 slot lives in half 0

  // ---------------- per-wave constant A operands (whole scan in VGPRs) ------
  // gate order i,f,g,o ; i/f/o pre-scaled by 0.5 for the tanh-based sigmoid
  v16h A[4];
  #pragma unroll
  for (int g = 0; g < 4; ++g) {
    const float scale = (g == 2) ? 1.0f : 0.5f;
    const int    m  = g * 16 + n;
    const float* wr = W_hh + (size_t)m * LSTM_H + 8 * half;
    v16h a = {};
    #pragma unroll
    for (int e = 0; e < 8; ++e) a[e] = (_Float16)(scale * wr[e]); // K = e+8*half
    if (half == 0) {                                       // K=16: W_ih, K=17: bias
      a[8] = (_Float16)(scale * W_ih[m]);                  // IN == 1 -> W_ih is [64,1]
      a[9] = (_Float16)(scale * (b_ih[m] + b_hh[m]));
    }
    A[g] = a;
  }

  // ---------------- state ---------------------------------------------------
  v8f  c  = {};                 // cell state c[hid=r+8*half][batch=n], f32
  v16h hB = {};                 // B operand: elems 0..7 = h (f16), 8 = x_t, 9 = 1.0
  hB[9] = (_Float16)xsel;       // constant 1.0 multiplies the bias row (half 0 only)

  const v8f czero = {};

  auto step = [&](float xv) {
    hB[8] = (_Float16)(xsel * xv);            // inject x_t at K=16 (v_fma_mixlo_f16)
    v8f acc[4];
    #pragma unroll
    for (int g = 0; g < 4; ++g) {
      acc[g] = __builtin_amdgcn_wmma_f32_16x16x32_f16(
          /*neg_a=*/false, A[g], /*neg_b=*/false, hB,
          /*c_mod=*/(short)0, czero, /*reuse_a=*/false, /*reuse_b=*/false);
    }
    #pragma unroll
    for (int r = 0; r < 8; ++r) {
      // acc for i/f/o is already z/2 (weights pre-scaled)
      float ig = fmaf(0.5f, tanh_fast(acc[0][r]), 0.5f);
      float fg = fmaf(0.5f, tanh_fast(acc[1][r]), 0.5f);
      float gg = tanh_fast(acc[2][r]);
      float og = fmaf(0.5f, tanh_fast(acc[3][r]), 0.5f);
      float cv = fmaf(fg, c[r], ig * gg);
      c[r] = cv;
      hB[r] = (_Float16)(og * tanh_fast(cv)); // fused mul+cvt (v_fma_mixlo_f16)
    }
  };

  // ---------------- time scan with 8-step register-chunk x prefetch ---------
  const float* xrow = x + (size_t)(bbase + n) * (size_t)T;  // IN == 1
  if ((T & 7) == 0) {
    auto load8 = [](float* dst, const float* p) {
      float4 a0 = ((const float4*)p)[0];
      float4 a1 = ((const float4*)p)[1];
      dst[0] = a0.x; dst[1] = a0.y; dst[2] = a0.z; dst[3] = a0.w;
      dst[4] = a1.x; dst[5] = a1.y; dst[6] = a1.z; dst[7] = a1.w;
    };
    float buf[8];
    load8(buf, xrow);
    for (int t0 = 0; t0 < T; t0 += 8) {
      float nbuf[8] = {0, 0, 0, 0, 0, 0, 0, 0};
      if (t0 + 8 < T) load8(nbuf, xrow + t0 + 8);   // issued ahead of use
      #pragma unroll
      for (int j = 0; j < 8; ++j) step(buf[j]);
      #pragma unroll
      for (int j = 0; j < 8; ++j) buf[j] = nbuf[j];
    }
  } else {
    for (int t = 0; t < T; ++t) step(xrow[t]);
  }

  // ---------------- final FC: out[b] = h_last . W_fc + b_fc -----------------
  // h_last reconstructed from the f16 B operand (same precision the
  // recurrence itself used).
  float part = 0.0f;
  #pragma unroll
  for (int r = 0; r < 8; ++r) part = fmaf((float)hB[r], W_fc[r + 8 * half], part);
  part += __shfl_xor(part, 16, 32);        // combine hid 0..7 with hid 8..15
  if (half == 0) out[bbase + n] = part + b_fc[0];
}

extern "C" void kernel_launch(void* const* d_in, const int* in_sizes, int n_in,
                              void* d_out, int out_size, void* d_ws, size_t ws_size,
                              hipStream_t stream) {
  const float* x    = (const float*)d_in[0];
  const float* W_ih = (const float*)d_in[1];
  const float* W_hh = (const float*)d_in[2];
  const float* b_ih = (const float*)d_in[3];
  const float* b_hh = (const float*)d_in[4];
  const float* W_fc = (const float*)d_in[5];
  const float* b_fc = (const float*)d_in[6];
  float* out = (float*)d_out;

  int B = out_size;                                  // output shape [B, 1]
  int T = (B > 0) ? (int)(in_sizes[0] / B) : 0;      // x is [B, T, 1]

  int nTiles = (B + 15) / 16;
  const int wavesPerBlock = 8;                       // 256 threads = 8 wave32
  int blocks = (nTiles + wavesPerBlock - 1) / wavesPerBlock;
  if (blocks < 1) blocks = 1;

  lstm16_wmma_kernel<<<dim3(blocks), dim3(32 * wavesPerBlock), 0, stream>>>(
      x, W_ih, W_hh, b_ih, b_hh, W_fc, b_fc, out, B, T);
}
